// SelfAttention_57140244906512
// MI455X (gfx1250) — matrix operand order repaired and checked by reading, not compiled, not run on hardware
//
#include <hip/hip_runtime.h>
#include <hip/hip_bf16.h>
#include <hip/hip_fp16.h>

typedef __attribute__((ext_vector_type(16))) _Float16 v16h;
typedef __attribute__((ext_vector_type(8)))  _Float16 v8h;
typedef __attribute__((ext_vector_type(8)))  float    v8f;
typedef __attribute__((__vector_size__(4 * sizeof(int)))) int v4i;

#define BM 128
#define BN 128
#define BK 32

#define AS1 __attribute__((address_space(1)))
#define AS3 __attribute__((address_space(3)))

#if defined(__has_builtin)
#if __has_builtin(__builtin_amdgcn_global_load_async_to_lds_b128) && \
    __has_builtin(__builtin_amdgcn_s_wait_asynccnt)
#define HAVE_ASYNC_LDS 1
#endif
#endif
#ifndef HAVE_ASYNC_LDS
#define HAVE_ASYNC_LDS 0
#endif

// ---------------------------------------------------------------- elementwise
__global__ __launch_bounds__(256) void cvt_f32_f16(const float* __restrict__ in,
                                                   _Float16* __restrict__ out, int nelem) {
  int i = blockIdx.x * blockDim.x + threadIdx.x;
  int stride = gridDim.x * blockDim.x;
  for (; i < nelem; i += stride) out[i] = (_Float16)in[i];
}

// K̂_c[k, 256*j + t] = sign[c][j] * K[k, 256*perm[c][j] + t]
__global__ __launch_bounds__(256) void build_khat(const _Float16* __restrict__ Kin,
                                                  _Float16* __restrict__ Khat,
                                                  int comp, int total) {
  const int perm[4][4] = {{0,1,2,3},{1,0,3,2},{2,3,0,1},{3,2,1,0}};
  const int sgn [4][4] = {{1,-1,-1,-1},{1,1,1,-1},{1,-1,1,1},{1,1,-1,1}};
  int i = blockIdx.x * blockDim.x + threadIdx.x;
  int stride = gridDim.x * blockDim.x;
  for (; i < total; i += stride) {
    int k = i >> 10;          // row
    int d = i & 1023;         // feature
    int j = d >> 8;           // quaternion chunk
    int t = d & 255;
    _Float16 v = Kin[(size_t)k * 1024 + perm[comp][j] * 256 + t];
    Khat[i] = (sgn[comp][j] < 0) ? (_Float16)(-v) : v;
  }
}

// ---------------------------------------------------------------- WMMA GEMM
// D[m,n] = sum_k A[m,k] * B[n,k]   (A: MxK row-major, B: NxK row-major)
// MODE 0: f32 store to Df[(m)*ldd + col_off + n]
// MODE 1: f16 store to Dh[(m)*ldd + n]
// MODE 2: f16 transposed store to Dh[(n)*ldd + m]
template <int MODE>
__global__ __launch_bounds__(256)
void gemm_nt_wmma(const _Float16* __restrict__ A, const _Float16* __restrict__ B,
                  float* __restrict__ Df, _Float16* __restrict__ Dh,
                  int K, int ldd, int col_off) {
  __shared__ _Float16 As[2][BM * BK];
  __shared__ _Float16 Bs[2][BN * BK];

  const int tid   = threadIdx.x;
  const int lane  = tid & 31;
  const int wid   = tid >> 5;          // 8 waves
  const int r     = lane & 15;
  const int h     = lane >> 4;
  const int waveM = (wid & 3) * 32;    // 4 x 32 rows
  const int waveN = (wid >> 2) * 64;   // 2 x 64 cols
  const int bm    = blockIdx.y * BM;
  const int bn    = blockIdx.x * BN;

  const v8f zero = {};
  v8f acc[2][4];
#pragma unroll
  for (int i = 0; i < 2; ++i)
#pragma unroll
    for (int j = 0; j < 4; ++j) acc[i][j] = zero;

  // Each thread moves one 16-half (32B) segment per tile per matrix.
  const int ldrow = tid >> 1;          // 0..127
  const int ldcol = (tid & 1) * 16;    // 0 or 16 halves
  const _Float16* gA = &A[(size_t)(bm + ldrow) * K + ldcol];
  const _Float16* gB = &B[(size_t)(bn + ldrow) * K + ldcol];
  const int ldsoff = ldrow * BK + ldcol;

  auto prefetch_tile = [&](int buf, int k0) {
#if HAVE_ASYNC_LDS
    __builtin_amdgcn_global_load_async_to_lds_b128(
        (AS1 v4i*)(AS1 void*)(const void*)(gA + k0),
        (AS3 v4i*)(void*)&As[buf][ldsoff], 0, 0);
    __builtin_amdgcn_global_load_async_to_lds_b128(
        (AS1 v4i*)(AS1 void*)(const void*)(gA + k0 + 8),
        (AS3 v4i*)(void*)&As[buf][ldsoff + 8], 0, 0);
    __builtin_amdgcn_global_load_async_to_lds_b128(
        (AS1 v4i*)(AS1 void*)(const void*)(gB + k0),
        (AS3 v4i*)(void*)&Bs[buf][ldsoff], 0, 0);
    __builtin_amdgcn_global_load_async_to_lds_b128(
        (AS1 v4i*)(AS1 void*)(const void*)(gB + k0 + 8),
        (AS3 v4i*)(void*)&Bs[buf][ldsoff + 8], 0, 0);
#else
    *(v16h*)&As[buf][ldsoff] = *(const v16h*)(gA + k0);
    *(v16h*)&Bs[buf][ldsoff] = *(const v16h*)(gB + k0);
#endif
  };

  prefetch_tile(0, 0);
  const int T = K / BK;
  for (int t = 0; t < T; ++t) {
    const int cur = t & 1;
#if HAVE_ASYNC_LDS
    __builtin_amdgcn_s_wait_asynccnt(0);
#endif
    __syncthreads();
    if (t + 1 < T) prefetch_tile(cur ^ 1, (t + 1) * BK);
    if (t + 2 < T) {  // warm L2 two tiles ahead
      __builtin_prefetch(gA + (t + 2) * BK, 0, 1);
      __builtin_prefetch(gB + (t + 2) * BK, 0, 1);
    }

    // A fragment (16x32 f16): lane r=row; halves[0:8] = K[h*16..h*16+7],
    // halves[8:16] = K[h*16+8 .. h*16+15]
    v16h af[2];
#pragma unroll
    for (int i = 0; i < 2; ++i) {
      const _Float16* p = &As[cur][(waveM + i * 16 + r) * BK];
      v8h lo = *(const v8h*)&p[h * 16];
      v8h hi = *(const v8h*)&p[h * 16 + 8];
      af[i] = __builtin_shufflevector(lo, hi, 0, 1, 2, 3, 4, 5, 6, 7,
                                      8, 9, 10, 11, 12, 13, 14, 15);
    }
    // B fragment (32x16 f16): lane r=col n; halves = K[h*16 .. h*16+15]
    v16h bf[4];
#pragma unroll
    for (int j = 0; j < 4; ++j) {
      const _Float16* p = &Bs[cur][(waveN + j * 16 + r) * BK];
      bf[j] = *(const v16h*)&p[h * 16];
    }
#pragma unroll
    for (int i = 0; i < 2; ++i)
#pragma unroll
      for (int j = 0; j < 4; ++j)
        acc[i][j] = __builtin_amdgcn_wmma_f32_16x16x32_f16(
            false, af[i], false, bf[j], (short)0, acc[i][j], false, false);
  }

  // C/D layout: VGPR v, lanes 0-15 -> M=v, lanes 16-31 -> M=v+8; N = r
#pragma unroll
  for (int i = 0; i < 2; ++i)
#pragma unroll
    for (int j = 0; j < 4; ++j)
#pragma unroll
      for (int v = 0; v < 8; ++v) {
        int gm = bm + waveM + i * 16 + h * 8 + v;
        int gn = bn + waveN + j * 16 + r;
        float val = acc[i][j][v];
        if (MODE == 0)      Df[(size_t)gm * ldd + col_off + gn] = val;
        else if (MODE == 1) Dh[(size_t)gm * ldd + gn] = (_Float16)val;
        else                Dh[(size_t)gn * ldd + gm] = (_Float16)val;
      }
}

// ---------------------------------------------------------------- softmax
// In-place f32 row softmax + f16 copy for the AV WMMA GEMM.
__global__ __launch_bounds__(256)
void softmax_row(float* __restrict__ L, _Float16* __restrict__ P16, int ncols) {
  __shared__ float red[256];
  const int row = blockIdx.x;
  float* rowp = L + (size_t)row * ncols;
  const int tid = threadIdx.x;

  float m = -3.402823466e38f;
  for (int c = tid; c < ncols; c += 256) m = fmaxf(m, rowp[c]);
  red[tid] = m; __syncthreads();
  for (int s = 128; s > 0; s >>= 1) {
    if (tid < s) red[tid] = fmaxf(red[tid], red[tid + s]);
    __syncthreads();
  }
  m = red[0]; __syncthreads();

  float sum = 0.f;
  for (int c = tid; c < ncols; c += 256) {
    float e = __expf(rowp[c] - m);
    rowp[c] = e;
    sum += e;
  }
  red[tid] = sum; __syncthreads();
  for (int s = 128; s > 0; s >>= 1) {
    if (tid < s) red[tid] += red[tid + s];
    __syncthreads();
  }
  float inv = 1.0f / red[0];

  for (int c = tid; c < ncols; c += 256) {
    float w = rowp[c] * inv;
    rowp[c] = w;
    P16[(size_t)row * ncols + c] = (_Float16)w;
  }
}

// ---------------------------------------------------------------- launcher
extern "C" void kernel_launch(void* const* d_in, const int* in_sizes, int n_in,
                              void* d_out, int out_size, void* d_ws, size_t ws_size,
                              hipStream_t stream) {
  const float* x  = (const float*)d_in[0];
  const float* WQ = (const float*)d_in[1];
  const float* WK = (const float*)d_in[2];
  const float* WV = (const float*)d_in[3];

  const int m = 1024;
  const int n = in_sizes[0] / m;          // 4096
  const int dq = m / 4;                   // 256

  float* y    = (float*)d_out;            // [n, m]
  float* wlog = y + (size_t)n * m;        // [n, n]: logits scratch; last = w output

  _Float16* ws  = (_Float16*)d_ws;
  _Float16* x16  = ws; ws += (size_t)n * m;
  _Float16* wq16 = ws; ws += (size_t)m * m;
  _Float16* wk16 = ws; ws += (size_t)m * m;
  _Float16* wv16 = ws; ws += (size_t)m * m;
  _Float16* q16  = ws; ws += (size_t)n * m;
  _Float16* k16  = ws; ws += (size_t)n * m;
  _Float16* vt16 = ws; ws += (size_t)n * m;   // V^T: [m, n]
  _Float16* kh16 = ws; ws += (size_t)n * m;
  _Float16* p16  = ws; ws += (size_t)n * n;   // f16 probs

  const dim3 blk(256);

  cvt_f32_f16<<<dim3(1024), blk, 0, stream>>>(x,  x16,  n * m);
  cvt_f32_f16<<<dim3(256),  blk, 0, stream>>>(WQ, wq16, m * m);
  cvt_f32_f16<<<dim3(256),  blk, 0, stream>>>(WK, wk16, m * m);
  cvt_f32_f16<<<dim3(256),  blk, 0, stream>>>(WV, wv16, m * m);

  // Projections: Q = x Wq^T, K = x Wk^T (f16), V^T stored transposed (f16)
  gemm_nt_wmma<1><<<dim3(m / BN, n / BM), blk, 0, stream>>>(x16, wq16, nullptr, q16,  m, m, 0);
  gemm_nt_wmma<1><<<dim3(m / BN, n / BM), blk, 0, stream>>>(x16, wk16, nullptr, k16,  m, m, 0);
  gemm_nt_wmma<2><<<dim3(m / BN, n / BM), blk, 0, stream>>>(x16, wv16, nullptr, vt16, m, n, 0);

  for (int c = 0; c < 4; ++c) {
    // K̂_c = signed chunk permutation of K
    build_khat<<<dim3(4096), blk, 0, stream>>>(k16, kh16, c, n * m);
    // logits_c = Q K̂_c^T  (f32 into w region of d_out)
    gemm_nt_wmma<0><<<dim3(n / BN, n / BM), blk, 0, stream>>>(q16, kh16, wlog, nullptr, m, n, 0);
    // softmax rows -> f32 in place (final w for c==3) + f16 probs
    softmax_row<<<dim3(n), blk, 0, stream>>>(wlog, p16, n);
    // y[:, c*256:(c+1)*256] = P V_c   (NT against V^T rows [c*256, c*256+256))
    gemm_nt_wmma<0><<<dim3(dq / BN, n / BM), blk, 0, stream>>>(
        p16, vt16 + (size_t)c * dq * n, y, nullptr, n, m, c * dq);
  }
}
